// LightningDiT_15710990368876
// MI455X (gfx1250) — matrix operand order repaired and checked by reading, not compiled, not run on hardware
//
#include <hip/hip_runtime.h>
#include <hip/hip_bf16.h>
#include <math.h>

// ---------------- model constants ----------------
constexpr int kHID   = 1152;
constexpr int kNH    = 16;
constexpr int kHD    = 72;     // head dim
constexpr int kHDP   = 96;     // padded head dim (3 x 32 for WMMA K)
constexpr int kB     = 2;
constexpr int kL     = 1024;   // tokens per image
constexpr int kM     = kB * kL;  // 2048 rows
constexpr int kDepth = 4;
constexpr int kMlp   = 3072;
constexpr int kFreq  = 256;

// ---------------- WMMA types ----------------
typedef __attribute__((ext_vector_type(16))) __bf16          v16bf;
typedef __attribute__((ext_vector_type(8)))  float           v8f;

struct Frag32 { uint4 lo; uint4 hi; };   // 32B == v16bf

__device__ __forceinline__ unsigned short f2bf(float f) {
  unsigned int u = __float_as_uint(f);
  u += 0x7FFFu + ((u >> 16) & 1u);       // round-to-nearest-even
  return (unsigned short)(u >> 16);
}
__device__ __forceinline__ float bf2f(unsigned short h) {
  return __uint_as_float(((unsigned int)h) << 16);
}
__device__ __forceinline__ float silu_f(float x) { return x / (1.0f + __expf(-x)); }

__device__ __forceinline__ v8f wmma_bf16(v16bf a, v16bf b, v8f c) {
  return __builtin_amdgcn_wmma_f32_16x16x32_bf16(false, a, false, b, (short)0, c, false, false);
}
// two contiguous 16B LDS/global loads -> one 32B fragment
__device__ __forceinline__ v16bf load_frag16(const unsigned short* p) {
  Frag32 f;
  f.lo = *reinterpret_cast<const uint4*>(p);
  f.hi = *reinterpret_cast<const uint4*>(p + 8);
  return __builtin_bit_cast(v16bf, f);
}
// A fragment: lane m fixed, k = 16*hsel + e
__device__ __forceinline__ v16bf load_fragA(const unsigned short* rowp, int hsel) {
  Frag32 f;
  f.lo = *reinterpret_cast<const uint4*>(rowp + 16 * hsel);
  f.hi = *reinterpret_cast<const uint4*>(rowp + 16 * hsel + 8);
  return __builtin_bit_cast(v16bf, f);
}
// scatter one row-major uint4 (8 halves along n) into a column-major LDS tile
__device__ __forceinline__ void store_t8(unsigned short* base, int stride, uint4 w) {
  base[0 * stride] = (unsigned short)(w.x);
  base[1 * stride] = (unsigned short)(w.x >> 16);
  base[2 * stride] = (unsigned short)(w.y);
  base[3 * stride] = (unsigned short)(w.y >> 16);
  base[4 * stride] = (unsigned short)(w.z);
  base[5 * stride] = (unsigned short)(w.z >> 16);
  base[6 * stride] = (unsigned short)(w.w);
  base[7 * stride] = (unsigned short)(w.w >> 16);
}

// =====================================================================
// Generic bf16 WMMA GEMM, double-buffered LDS pipeline (one barrier/K-step).
// Block tile 128x128, BK=32, 256 threads = 8 waves (wave32), wave tile 32x64.
// mode 0: outf = acc + bias; mode 1: outh = bf16(acc + bias);
// mode 2: resid += gate[b][col] * (acc + bias)
// =====================================================================
constexpr int BM = 128, BN = 128, BK = 32;
constexpr int BTS = 40;   // padded LDS stride (halves) for transposed B tile

__global__ __launch_bounds__(256) void gemm_bf16_kernel(
    const unsigned short* __restrict__ A,
    const unsigned short* __restrict__ Bw,
    const float* __restrict__ bias,
    float* __restrict__ outf,
    unsigned short* __restrict__ outh,
    const float* __restrict__ gate, int gstride,
    float* __restrict__ resid,
    int N, int K, int mode)
{
  __shared__ unsigned short sA [2][BM * BK];    // row-major    [m][k]
  __shared__ unsigned short sBt[2][BN * BTS];   // column-major [n][k]

  const int tid  = threadIdx.x;
  const int lane = tid & 31;
  const int wave = tid >> 5;
  const int wm   = wave >> 1;
  const int wn   = wave & 1;
  const int hsel = lane >> 4;
  const int ln   = lane & 15;
  const int m0   = blockIdx.y * BM;
  const int n0   = blockIdx.x * BN;

  const int aRow0 = (tid * 2) >> 2,     aCc0 = (tid * 2) & 3;
  const int aRow1 = (tid * 2 + 1) >> 2, aCc1 = (tid * 2 + 1) & 3;
  const int bK0   = (tid * 2) & 31,     bCc0 = (tid * 2) >> 5;
  const int bK1   = (tid * 2 + 1) & 31, bCc1 = (tid * 2 + 1) >> 5;

  v8f acc[2][4];
  v8f zero = {0.f,0.f,0.f,0.f,0.f,0.f,0.f,0.f};
  #pragma unroll
  for (int i = 0; i < 2; ++i)
    #pragma unroll
    for (int j = 0; j < 4; ++j) acc[i][j] = zero;

  uint4 aR0, aR1, bR0, bR1;
  aR0 = *reinterpret_cast<const uint4*>(&A [(size_t)(m0 + aRow0) * K + aCc0 * 8]);
  aR1 = *reinterpret_cast<const uint4*>(&A [(size_t)(m0 + aRow1) * K + aCc1 * 8]);
  bR0 = *reinterpret_cast<const uint4*>(&Bw[(size_t)(bK0)       * N + n0 + bCc0 * 8]);
  bR1 = *reinterpret_cast<const uint4*>(&Bw[(size_t)(bK1)       * N + n0 + bCc1 * 8]);
  *reinterpret_cast<uint4*>(&sA[0][aRow0 * BK + aCc0 * 8]) = aR0;
  *reinterpret_cast<uint4*>(&sA[0][aRow1 * BK + aCc1 * 8]) = aR1;
  store_t8(&sBt[0][(bCc0 * 8) * BTS + bK0], BTS, bR0);
  store_t8(&sBt[0][(bCc1 * 8) * BTS + bK1], BTS, bR1);
  __syncthreads();

  const int nk = K / BK;
  int cur = 0;
  for (int it = 0; it < nk; ++it) {
    const bool more = (it + 1 < nk);
    if (more) {   // next-tile global loads in flight during WMMAs
      const int kt = (it + 1) * BK;
      aR0 = *reinterpret_cast<const uint4*>(&A [(size_t)(m0 + aRow0) * K + kt + aCc0 * 8]);
      aR1 = *reinterpret_cast<const uint4*>(&A [(size_t)(m0 + aRow1) * K + kt + aCc1 * 8]);
      bR0 = *reinterpret_cast<const uint4*>(&Bw[(size_t)(kt + bK0) * N + n0 + bCc0 * 8]);
      bR1 = *reinterpret_cast<const uint4*>(&Bw[(size_t)(kt + bK1) * N + n0 + bCc1 * 8]);
    }

    const unsigned short* sAc = sA[cur];
    const unsigned short* sBc = sBt[cur];
    v16bf af0 = load_fragA(&sAc[(wm * 32 +  0 + ln) * BK], hsel);
    v16bf af1 = load_fragA(&sAc[(wm * 32 + 16 + ln) * BK], hsel);
    v16bf bf0 = load_frag16(&sBc[(wn * 64 +  0 + ln) * BTS + 16 * hsel]);
    v16bf bf1 = load_frag16(&sBc[(wn * 64 + 16 + ln) * BTS + 16 * hsel]);
    v16bf bf2 = load_frag16(&sBc[(wn * 64 + 32 + ln) * BTS + 16 * hsel]);
    v16bf bf3 = load_frag16(&sBc[(wn * 64 + 48 + ln) * BTS + 16 * hsel]);
    acc[0][0] = wmma_bf16(af0, bf0, acc[0][0]);
    acc[1][0] = wmma_bf16(af1, bf0, acc[1][0]);
    acc[0][1] = wmma_bf16(af0, bf1, acc[0][1]);
    acc[1][1] = wmma_bf16(af1, bf1, acc[1][1]);
    acc[0][2] = wmma_bf16(af0, bf2, acc[0][2]);
    acc[1][2] = wmma_bf16(af1, bf2, acc[1][2]);
    acc[0][3] = wmma_bf16(af0, bf3, acc[0][3]);
    acc[1][3] = wmma_bf16(af1, bf3, acc[1][3]);

    if (more) {
      const int nxt = cur ^ 1;
      *reinterpret_cast<uint4*>(&sA[nxt][aRow0 * BK + aCc0 * 8]) = aR0;
      *reinterpret_cast<uint4*>(&sA[nxt][aRow1 * BK + aCc1 * 8]) = aR1;
      store_t8(&sBt[nxt][(bCc0 * 8) * BTS + bK0], BTS, bR0);
      store_t8(&sBt[nxt][(bCc1 * 8) * BTS + bK1], BTS, bR1);
    }
    __syncthreads();
    cur ^= 1;
  }

  #pragma unroll
  for (int i = 0; i < 2; ++i)
    #pragma unroll
    for (int j = 0; j < 4; ++j)
      #pragma unroll
      for (int r = 0; r < 8; ++r) {
        int row = m0 + wm * 32 + i * 16 + r + 8 * hsel;
        int col = n0 + wn * 64 + j * 16 + ln;
        float v = acc[i][j][r] + bias[col];
        if (mode == 0) {
          outf[(size_t)row * N + col] = v;
        } else if (mode == 1) {
          outh[(size_t)row * N + col] = f2bf(v);
        } else {
          int b = row >> 10;
          resid[(size_t)row * N + col] += gate[b * gstride + col] * v;
        }
      }
}

// =====================================================================
// Fused flash attention, double-buffered K/V tiles.
// =====================================================================
constexpr int VTS = 40;

__global__ __launch_bounds__(128) void attn_kernel(
    const unsigned short* __restrict__ qp,
    const unsigned short* __restrict__ kp,
    const unsigned short* __restrict__ vp,
    float* __restrict__ outp)
{
  __shared__ unsigned short sk [2][32 * kHDP];
  __shared__ unsigned short svt[2][kHDP * VTS];
  __shared__ unsigned short sp[4][16 * 32];

  const int tid  = threadIdx.x;
  const int lane = tid & 31;
  const int wave = tid >> 5;
  const int hsel = lane >> 4;
  const int ln   = lane & 15;
  const int bh   = blockIdx.x >> 4;
  const int qc   = blockIdx.x & 15;
  const int b    = bh >> 4;
  const int hh   = bh & 15;
  const int qrow0 = qc * 64 + wave * 16;
  const size_t headbase = (size_t)bh * kL * kHDP;

  v16bf qf[3];
  {
    const unsigned short* qptr = qp + headbase + (size_t)(qrow0 + ln) * kHDP;
    #pragma unroll
    for (int kc = 0; kc < 3; ++kc)
      qf[kc] = load_fragA(qptr + kc * 32, hsel);
  }

  float mrow[8], lrow[8];
  v8f oacc[5];
  v8f zero = {0.f,0.f,0.f,0.f,0.f,0.f,0.f,0.f};
  #pragma unroll
  for (int r = 0; r < 8; ++r) { mrow[r] = -1e30f; lrow[r] = 0.f; }
  #pragma unroll
  for (int t = 0; t < 5; ++t) oacc[t] = zero;

  int skey[3], scc[3];
  #pragma unroll
  for (int i = 0; i < 3; ++i) { int c = tid + i * 128; skey[i] = c & 31; scc[i] = c >> 5; }

  uint4 kReg[3], vReg[3];
  #pragma unroll
  for (int i = 0; i < 3; ++i) {
    const size_t gsrc = headbase + (size_t)(skey[i]) * kHDP + scc[i] * 8;
    kReg[i] = *reinterpret_cast<const uint4*>(kp + gsrc);
    vReg[i] = *reinterpret_cast<const uint4*>(vp + gsrc);
  }
  #pragma unroll
  for (int i = 0; i < 3; ++i) {
    *reinterpret_cast<uint4*>(&sk[0][skey[i] * kHDP + scc[i] * 8]) = kReg[i];
    store_t8(&svt[0][(scc[i] * 8) * VTS + skey[i]], VTS, vReg[i]);
  }
  __syncthreads();

  int cur = 0;
  for (int kt = 0; kt < kL; kt += 32) {
    const bool more = (kt + 32 < kL);
    if (more) {
      #pragma unroll
      for (int i = 0; i < 3; ++i) {
        const size_t gsrc = headbase + (size_t)(kt + 32 + skey[i]) * kHDP + scc[i] * 8;
        kReg[i] = *reinterpret_cast<const uint4*>(kp + gsrc);
        vReg[i] = *reinterpret_cast<const uint4*>(vp + gsrc);
      }
    }
    const unsigned short* skc  = sk[cur];
    const unsigned short* svtc = svt[cur];

    v8f s0 = zero, s1 = zero;
    #pragma unroll
    for (int kc = 0; kc < 3; ++kc) {
      v16bf b0 = load_frag16(&skc[(ln)      * kHDP + kc * 32 + 16 * hsel]);
      v16bf b1 = load_frag16(&skc[(16 + ln) * kHDP + kc * 32 + 16 * hsel]);
      s0 = wmma_bf16(qf[kc], b0, s0);
      s1 = wmma_bf16(qf[kc], b1, s1);
    }

    #pragma unroll
    for (int r = 0; r < 8; ++r) {
      float x0 = s0[r], x1 = s1[r];
      float mx = fmaxf(x0, x1);
      #pragma unroll
      for (int msk = 8; msk >= 1; msk >>= 1) mx = fmaxf(mx, __shfl_xor(mx, msk, 32));
      float mnew = fmaxf(mrow[r], mx);
      float corr = __expf(mrow[r] - mnew);
      float p0 = __expf(x0 - mnew);
      float p1 = __expf(x1 - mnew);
      float rs = p0 + p1;
      #pragma unroll
      for (int msk = 8; msk >= 1; msk >>= 1) rs += __shfl_xor(rs, msk, 32);
      lrow[r] = lrow[r] * corr + rs;
      mrow[r] = mnew;
      #pragma unroll
      for (int t = 0; t < 5; ++t) oacc[t][r] *= corr;
      int mr = r + 8 * hsel;
      sp[wave][mr * 32 + ln]      = f2bf(p0);
      sp[wave][mr * 32 + 16 + ln] = f2bf(p1);
    }

    v16bf pa = load_fragA(&sp[wave][ln * 32], hsel);
    v16bf v0 = load_frag16(&svtc[( 0 + ln) * VTS + 16 * hsel]);
    v16bf v1 = load_frag16(&svtc[(16 + ln) * VTS + 16 * hsel]);
    v16bf v2 = load_frag16(&svtc[(32 + ln) * VTS + 16 * hsel]);
    v16bf v3 = load_frag16(&svtc[(48 + ln) * VTS + 16 * hsel]);
    v16bf v4 = load_frag16(&svtc[(64 + ln) * VTS + 16 * hsel]);
    oacc[0] = wmma_bf16(pa, v0, oacc[0]);
    oacc[1] = wmma_bf16(pa, v1, oacc[1]);
    oacc[2] = wmma_bf16(pa, v2, oacc[2]);
    oacc[3] = wmma_bf16(pa, v3, oacc[3]);
    oacc[4] = wmma_bf16(pa, v4, oacc[4]);

    if (more) {
      const int nxt = cur ^ 1;
      #pragma unroll
      for (int i = 0; i < 3; ++i) {
        *reinterpret_cast<uint4*>(&sk[nxt][skey[i] * kHDP + scc[i] * 8]) = kReg[i];
        store_t8(&svt[nxt][(scc[i] * 8) * VTS + skey[i]], VTS, vReg[i]);
      }
    }
    __syncthreads();
    cur ^= 1;
  }

  #pragma unroll
  for (int t = 0; t < 5; ++t) {
    int col = t * 16 + ln;
    if (col < kHD) {
      #pragma unroll
      for (int r = 0; r < 8; ++r) {
        int row = qrow0 + r + 8 * hsel;
        outp[((size_t)b * kL + row) * kHID + hh * kHD + col] = oacc[t][r] / lrow[r];
      }
    }
  }
}

// =====================================================================
// small / elementwise kernels
// =====================================================================
__global__ void f2bf_kernel(const float* __restrict__ in, unsigned short* __restrict__ out, int n)
{
  int i = blockIdx.x * blockDim.x + threadIdx.x;
  if (i < n) out[i] = f2bf(in[i]);
}

__global__ void patch_kernel(const float* __restrict__ x, const float* __restrict__ pw,
                             const float* __restrict__ pb, float* __restrict__ h)
{
  int idx = blockIdx.x * blockDim.x + threadIdx.x;
  if (idx >= kM * kHID) return;
  int col = idx % kHID, row = idx / kHID;
  int b = row >> 10, l = row & 1023;
  int sy = l >> 5, sx = l & 31;
  float acc = pb[col];
  #pragma unroll
  for (int kk = 0; kk < 16; ++kk) {
    int c = kk & 3, px = (kk >> 2) & 1, py = kk >> 3;
    float xv = x[(((size_t)b * 4 + c) * 64 + sy * 2 + py) * 64 + sx * 2 + px];
    acc += xv * pw[kk * kHID + col];
  }
  h[(size_t)row * kHID + col] = acc;
}

__global__ __launch_bounds__(128) void emb_kernel(
    const float* __restrict__ t, const int* __restrict__ y,
    const float* __restrict__ w1, const float* __restrict__ b1,
    const float* __restrict__ w2, const float* __restrict__ b2,
    const float* __restrict__ ytab,
    float* __restrict__ cbuf, float* __restrict__ csilu)
{
  __shared__ float te[kFreq];
  __shared__ float t1s[kHID];
  int b = blockIdx.x;
  float tv = t[b];
  for (int i = threadIdx.x; i < kFreq / 2; i += 128) {
    float fr = __expf(-logf(10000.f) * (float)i / (float)(kFreq / 2));
    float a = tv * fr;
    te[i] = __cosf(a);
    te[kFreq / 2 + i] = __sinf(a);
  }
  __syncthreads();
  for (int c = threadIdx.x; c < kHID; c += 128) {
    float acc = b1[c];
    for (int k = 0; k < kFreq; ++k) acc += te[k] * w1[k * kHID + c];
    t1s[c] = silu_f(acc);
  }
  __syncthreads();
  int yb = y[b];
  for (int c = threadIdx.x; c < kHID; c += 128) {
    float acc = b2[c];
    for (int k = 0; k < kHID; ++k) acc += t1s[k] * w2[(size_t)k * kHID + c];
    acc += ytab[(size_t)yb * kHID + c];
    cbuf[b * kHID + c]  = acc;
    csilu[b * kHID + c] = silu_f(acc);
  }
}

__global__ void mod_kernel(const float* __restrict__ csilu, const float* __restrict__ aw,
                           const float* __restrict__ ab, float* __restrict__ modb)
{
  int idx = blockIdx.x * blockDim.x + threadIdx.x;
  if (idx >= kDepth * kB * 4 * kHID) return;
  int col = idx % (4 * kHID);
  int rem = idx / (4 * kHID);
  int b = rem % kB, layer = rem / kB;
  float acc = ab[layer * 4 * kHID + col];
  const float* cs = csilu + b * kHID;
  const float* w  = aw + (size_t)layer * kHID * 4 * kHID + col;
  for (int k = 0; k < kHID; ++k) acc += cs[k] * w[(size_t)k * 4 * kHID];
  modb[((size_t)layer * kB + b) * (4 * kHID) + col] = acc;
}

__global__ void fmod_kernel(const float* __restrict__ csilu, const float* __restrict__ w,
                            const float* __restrict__ bias, float* __restrict__ fm)
{
  int idx = blockIdx.x * blockDim.x + threadIdx.x;
  if (idx >= kB * 2 * kHID) return;
  int col = idx % (2 * kHID), b = idx / (2 * kHID);
  float acc = bias[col];
  for (int k = 0; k < kHID; ++k) acc += csilu[b * kHID + k] * w[(size_t)k * 2 * kHID + col];
  fm[idx] = acc;
}

__global__ void rope_tab_kernel(float* __restrict__ cosb, float* __restrict__ sinb)
{
  int idx = blockIdx.x * blockDim.x + threadIdx.x;
  if (idx >= kL * kHD) return;
  int d = idx % kHD, l = idx / kHD;
  int sy = l >> 5, sx = l & 31;
  int pos = (d < 36) ? sy : sx;
  int j = (d < 36) ? d : d - 36;
  int i = j >> 1;
  float f = __powf(10000.f, -(float)(2 * i) / 36.f);
  float a = (float)pos * f;
  cosb[idx] = __cosf(a);
  sinb[idx] = __sinf(a);
}

__global__ void rope_pack_kernel(const float* __restrict__ qkvf,
                                 const float* __restrict__ cosb, const float* __restrict__ sinb,
                                 unsigned short* __restrict__ qp, unsigned short* __restrict__ kp,
                                 unsigned short* __restrict__ vp)
{
  int idx = blockIdx.x * blockDim.x + threadIdx.x;
  if (idx >= kB * kNH * kL * 48) return;
  int pr  = idx % 48;
  int rem = idx / 48;
  int l   = rem % kL;
  int bh  = rem / kL;
  int hh = bh % kNH, b = bh / kNH;
  size_t dst = ((size_t)bh * kL + l) * kHDP;
  const float scale = 0.117851130198f;   // 1/sqrt(72)
  if (pr < 36) {
    size_t src = ((size_t)b * kL + l) * (3 * kHID) + hh * kHD + 2 * pr;
    float c = cosb[l * kHD + 2 * pr], s = sinb[l * kHD + 2 * pr];
    float q0 = qkvf[src],            q1 = qkvf[src + 1];
    float k0 = qkvf[src + kHID],     k1 = qkvf[src + kHID + 1];
    float v0 = qkvf[src + 2 * kHID], v1 = qkvf[src + 2 * kHID + 1];
    qp[dst + 2 * pr]     = f2bf((q0 * c - q1 * s) * scale);
    qp[dst + 2 * pr + 1] = f2bf((q1 * c + q0 * s) * scale);
    kp[dst + 2 * pr]     = f2bf(k0 * c - k1 * s);
    kp[dst + 2 * pr + 1] = f2bf(k1 * c + k0 * s);
    vp[dst + 2 * pr]     = f2bf(v0);
    vp[dst + 2 * pr + 1] = f2bf(v1);
  } else {
    int d = kHD + (pr - 36) * 2;
    qp[dst + d] = 0; qp[dst + d + 1] = 0;
    kp[dst + d] = 0; kp[dst + d + 1] = 0;
    vp[dst + d] = 0; vp[dst + d + 1] = 0;
  }
}

__global__ __launch_bounds__(128) void rmsnorm_mod_kernel(
    const float* __restrict__ h, const float* __restrict__ nw,
    const float* __restrict__ smod, int sstride, unsigned short* __restrict__ out)
{
  __shared__ float red[4];
  int row = blockIdx.x;
  int b = row >> 10;
  const float* hp = h + (size_t)row * kHID;
  float ss = 0.f;
  for (int c = threadIdx.x; c < kHID; c += 128) { float v = hp[c]; ss += v * v; }
  #pragma unroll
  for (int msk = 16; msk >= 1; msk >>= 1) ss += __shfl_xor(ss, msk, 32);
  if ((threadIdx.x & 31) == 0) red[threadIdx.x >> 5] = ss;
  __syncthreads();
  ss = red[0] + red[1] + red[2] + red[3];
  float r = rsqrtf(ss / (float)kHID + 1e-6f);
  for (int c = threadIdx.x; c < kHID; c += 128) {
    float v = hp[c] * r * nw[c] * (1.f + smod[b * sstride + c]);
    out[(size_t)row * kHID + c] = f2bf(v);
  }
}

__global__ __launch_bounds__(128) void final_norm_kernel(
    const float* __restrict__ h, const float* __restrict__ nw,
    const float* __restrict__ fm, float* __restrict__ out)
{
  __shared__ float red[4];
  int row = blockIdx.x;
  int b = row >> 10;
  const float* hp = h + (size_t)row * kHID;
  float ss = 0.f;
  for (int c = threadIdx.x; c < kHID; c += 128) { float v = hp[c]; ss += v * v; }
  #pragma unroll
  for (int msk = 16; msk >= 1; msk >>= 1) ss += __shfl_xor(ss, msk, 32);
  if ((threadIdx.x & 31) == 0) red[threadIdx.x >> 5] = ss;
  __syncthreads();
  ss = red[0] + red[1] + red[2] + red[3];
  float r = rsqrtf(ss / (float)kHID + 1e-6f);
  for (int c = threadIdx.x; c < kHID; c += 128) {
    float v = hp[c] * r * nw[c] * (1.f + fm[b * 2 * kHID + kHID + c]) + fm[b * 2 * kHID + c];
    out[(size_t)row * kHID + c] = v;
  }
}

__global__ void swiglu_kernel(const unsigned short* __restrict__ x12, unsigned short* __restrict__ m16)
{
  int idx = blockIdx.x * blockDim.x + threadIdx.x;
  if (idx >= kM * kMlp) return;
  int i = idx % kMlp, row = idx / kMlp;
  float x1 = bf2f(x12[(size_t)row * 2 * kMlp + i]);
  float x2 = bf2f(x12[(size_t)row * 2 * kMlp + kMlp + i]);
  m16[idx] = f2bf(silu_f(x1) * x2);
}

__global__ void final_kernel(const float* __restrict__ hf, const float* __restrict__ w,
                             const float* __restrict__ bfl, float* __restrict__ out)
{
  int idx = blockIdx.x * blockDim.x + threadIdx.x;
  if (idx >= kM * 16) return;
  int col = idx & 15, row = idx >> 4;
  float acc = bfl[col];
  const float* hp = hf + (size_t)row * kHID;
  for (int k = 0; k < kHID; ++k) acc += hp[k] * w[k * 16 + col];
  int b = row >> 10, l = row & 1023;
  int sy = l >> 5, sx = l & 31;
  int c = col & 3, px = (col >> 2) & 1, py = col >> 3;
  out[(((size_t)b * 4 + c) * 64 + sy * 2 + py) * 64 + sx * 2 + px] = acc;
}

// =====================================================================
// host launch
// =====================================================================
extern "C" void kernel_launch(void* const* d_in, const int* in_sizes, int n_in,
                              void* d_out, int out_size, void* d_ws, size_t ws_size,
                              hipStream_t stream)
{
  (void)in_sizes; (void)n_in; (void)out_size; (void)ws_size;
  const float* x        = (const float*)d_in[0];
  const float* t        = (const float*)d_in[1];
  const int*   y        = (const int*)  d_in[2];
  const float* patch_w  = (const float*)d_in[3];
  const float* patch_b  = (const float*)d_in[4];
  const float* t_w1     = (const float*)d_in[5];
  const float* t_b1     = (const float*)d_in[6];
  const float* t_w2     = (const float*)d_in[7];
  const float* t_b2     = (const float*)d_in[8];
  const float* y_table  = (const float*)d_in[9];
  const float* norm1_w  = (const float*)d_in[10];
  const float* qkv_w    = (const float*)d_in[11];
  const float* qkv_b    = (const float*)d_in[12];
  const float* out_w    = (const float*)d_in[13];
  const float* out_b    = (const float*)d_in[14];
  const float* norm2_w  = (const float*)d_in[15];
  const float* mlp_w12  = (const float*)d_in[16];
  const float* mlp_b12  = (const float*)d_in[17];
  const float* mlp_w3   = (const float*)d_in[18];
  const float* mlp_b3   = (const float*)d_in[19];
  const float* adaln_w  = (const float*)d_in[20];
  const float* adaln_b  = (const float*)d_in[21];
  const float* fnorm_w  = (const float*)d_in[22];
  const float* fadaln_w = (const float*)d_in[23];
  const float* fadaln_b = (const float*)d_in[24];
  const float* flin_w   = (const float*)d_in[25];
  const float* flin_b   = (const float*)d_in[26];

  char* ws = (char*)d_ws;
  size_t off = 0;
  auto carve = [&](size_t bytes) -> void* {
    void* p = ws + off;
    off += (bytes + 255) & ~(size_t)255;
    return p;
  };
  float*          cosb   = (float*)carve((size_t)kL * kHD * 4);
  float*          sinb   = (float*)carve((size_t)kL * kHD * 4);
  float*          cbuf   = (float*)carve((size_t)kB * kHID * 4);
  float*          csilu  = (float*)carve((size_t)kB * kHID * 4);
  float*          modb   = (float*)carve((size_t)kDepth * kB * 4 * kHID * 4);
  float*          fmodb  = (float*)carve((size_t)kB * 2 * kHID * 4);
  float*          hbuf   = (float*)carve((size_t)kM * kHID * 4);
  unsigned short* act16  = (unsigned short*)carve((size_t)kM * kHID * 2);
  unsigned short* wconv  = (unsigned short*)carve((size_t)kHID * 2 * kMlp * 2);
  float*          qkvf   = (float*)carve((size_t)kM * 3 * kHID * 4);
  unsigned short* qpk    = (unsigned short*)carve((size_t)kB * kNH * kL * kHDP * 2);
  unsigned short* kpk    = (unsigned short*)carve((size_t)kB * kNH * kL * kHDP * 2);
  unsigned short* vpk    = (unsigned short*)carve((size_t)kB * kNH * kL * kHDP * 2);
  float*          attnout= (float*)carve((size_t)kM * kHID * 4);
  unsigned short* x12h   = (unsigned short*)carve((size_t)kM * 2 * kMlp * 2);
  unsigned short* m16    = (unsigned short*)carve((size_t)kM * kMlp * 2);
  float*          hf     = (float*)carve((size_t)kM * kHID * 4);

  emb_kernel<<<kB, 128, 0, stream>>>(t, y, t_w1, t_b1, t_w2, t_b2, y_table, cbuf, csilu);
  mod_kernel<<<(kDepth * kB * 4 * kHID + 255) / 256, 256, 0, stream>>>(csilu, adaln_w, adaln_b, modb);
  fmod_kernel<<<(kB * 2 * kHID + 255) / 256, 256, 0, stream>>>(csilu, fadaln_w, fadaln_b, fmodb);
  rope_tab_kernel<<<(kL * kHD + 255) / 256, 256, 0, stream>>>(cosb, sinb);
  patch_kernel<<<(kM * kHID + 255) / 256, 256, 0, stream>>>(x, patch_w, patch_b, hbuf);

  for (int layer = 0; layer < kDepth; ++layer) {
    const float* mod_l = modb + (size_t)layer * kB * 4 * kHID;

    rmsnorm_mod_kernel<<<kM, 128, 0, stream>>>(hbuf, norm1_w + layer * kHID,
                                               mod_l + 0 * kHID, 4 * kHID, act16);
    int nqw = kHID * 3 * kHID;
    f2bf_kernel<<<(nqw + 255) / 256, 256, 0, stream>>>(qkv_w + (size_t)layer * nqw, wconv, nqw);
    dim3 gq(3 * kHID / BN, kM / BM);
    gemm_bf16_kernel<<<gq, 256, 0, stream>>>(act16, wconv, qkv_b + layer * 3 * kHID,
                                             qkvf, nullptr, nullptr, 0, nullptr,
                                             3 * kHID, kHID, 0);
    int nrp = kB * kNH * kL * 48;
    rope_pack_kernel<<<(nrp + 255) / 256, 256, 0, stream>>>(qkvf, cosb, sinb, qpk, kpk, vpk);
    attn_kernel<<<kB * kNH * (kL / 64), 128, 0, stream>>>(qpk, kpk, vpk, attnout);
    int nao = kM * kHID;
    f2bf_kernel<<<(nao + 255) / 256, 256, 0, stream>>>(attnout, act16, nao);
    int now = kHID * kHID;
    f2bf_kernel<<<(now + 255) / 256, 256, 0, stream>>>(out_w + (size_t)layer * now, wconv, now);
    dim3 go(kHID / BN, kM / BM);
    gemm_bf16_kernel<<<go, 256, 0, stream>>>(act16, wconv, out_b + layer * kHID,
                                             nullptr, nullptr, mod_l + 1 * kHID, 4 * kHID, hbuf,
                                             kHID, kHID, 2);

    rmsnorm_mod_kernel<<<kM, 128, 0, stream>>>(hbuf, norm2_w + layer * kHID,
                                               mod_l + 2 * kHID, 4 * kHID, act16);
    int n12 = kHID * 2 * kMlp;
    f2bf_kernel<<<(n12 + 255) / 256, 256, 0, stream>>>(mlp_w12 + (size_t)layer * n12, wconv, n12);
    dim3 gm(2 * kMlp / BN, kM / BM);
    gemm_bf16_kernel<<<gm, 256, 0, stream>>>(act16, wconv, mlp_b12 + layer * 2 * kMlp,
                                             nullptr, x12h, nullptr, 0, nullptr,
                                             2 * kMlp, kHID, 1);
    int nsw = kM * kMlp;
    swiglu_kernel<<<(nsw + 255) / 256, 256, 0, stream>>>(x12h, m16);
    int n3 = kMlp * kHID;
    f2bf_kernel<<<(n3 + 255) / 256, 256, 0, stream>>>(mlp_w3 + (size_t)layer * n3, wconv, n3);
    gemm_bf16_kernel<<<go, 256, 0, stream>>>(m16, wconv, mlp_b3 + layer * kHID,
                                             nullptr, nullptr, mod_l + 3 * kHID, 4 * kHID, hbuf,
                                             kHID, kMlp, 2);
  }

  final_norm_kernel<<<kM, 128, 0, stream>>>(hbuf, fnorm_w, fmodb, hf);
  final_kernel<<<(kM * 16 + 255) / 256, 256, 0, stream>>>(hf, flin_w, flin_b, (float*)d_out);
}
